// ProtoNetLayer_8881992368372
// MI455X (gfx1250) — compile-verified, matrix operand order checked by reading, not run on hardware
//
#include <hip/hip_runtime.h>

typedef __attribute__((ext_vector_type(16))) _Float16 v16h;
typedef __attribute__((ext_vector_type(8)))  _Float16 v8h;
typedef __attribute__((ext_vector_type(4)))  _Float16 v4h;
typedef __attribute__((ext_vector_type(8)))  float    v8f;
typedef __attribute__((ext_vector_type(4)))  unsigned int v4u;
typedef __attribute__((ext_vector_type(8)))  int      v8i;
typedef __attribute__((ext_vector_type(4)))  int      v4i;

#define SHUF16(lo, hi) __builtin_shufflevector((lo), (hi), 0,1,2,3,4,5,6,7,8,9,10,11,12,13,14,15)

// Problem dims (fixed by the reference): D_in=1024, D_proj=256, C=256.
#define D_IN   1024
#define DPROJ  256
#define NCLS   256
#define BM     64      // rows per block
#define LDA    40      // A tile LDS stride (32 + 8 pad), elements
#define LDB    40      // W^T / P tile LDS stride (32 + 8 pad), elements
#define LZS    264     // normalized-Z LDS stride (256 + 8 pad), elements
#define EPS    1e-12f

// Dynamic-LDS carve (bytes)
#define OFF_A    0
#define OFF_BT0  5120          // 64*40*2
#define OFF_BT1  25600         // +256*40*2
#define OFF_ZS   46080         // +256*40*2
#define OFF_RS   79872         // +64*264*2
#define OFF_P2   80128         // +64*4
#define SMEM_BYTES 81152       // +256*4

#if defined(__has_builtin)
# if __has_builtin(__builtin_amdgcn_tensor_load_to_lds) && __has_builtin(__builtin_amdgcn_s_wait_tensorcnt)
#  define USE_TDM 1
# endif
#endif
#ifndef USE_TDM
# define USE_TDM 0
#endif

__device__ __forceinline__ unsigned lds_addr_of(const void* p) {
    return (unsigned)(uintptr_t)p;   // flat LDS aperture: addr[31:0] is the LDS byte address
}

#if USE_TDM
// TDM: load a [256 rows x 32 f16] tile, row pitch `stride0` elements, into LDS
// with +16B pad after each 64B row (matches LDS stride of 40 f16 = 80B).
// 6-arg builtin form (clang-23 / therock-10.0 headers):
//   (uint32x4 g0, int32x8 g1, int32x4 g2, int32x4 g3, int32x8 extra, i32 cpol)
__device__ __forceinline__ void tdm_load_tile(unsigned lds_byte_addr, const void* gtile,
                                              int tensor_dim0, int stride0) {
    unsigned long long ga = (unsigned long long)(uintptr_t)gtile;
    v4u g0;
    g0[0] = 1u;                                             // count=1, user descriptor
    g0[1] = lds_byte_addr;                                  // lds_addr
    g0[2] = (unsigned)(ga & 0xFFFFFFFFull);                 // global_addr[31:0]
    g0[3] = (unsigned)((ga >> 32) & 0x1FFFFFFull)           // global_addr[56:32]
          | 0x80000000u;                                    // type=2 ("image")
    v8i g1;
    g1[0] = 0x06D10000;  // data_size=2B | pad_enable | pad_interval=16DW | pad_amount=4DW
    g1[1] = (int)((unsigned)tensor_dim0 << 16);             // tensor_dim0[15:0]
    g1[2] = (int)((((unsigned)tensor_dim0) >> 16) | (256u << 16)); // dim0 hi | tensor_dim1=256
    g1[3] = (int)(32u << 16);                               // tile_dim0 = 32 elements
    g1[4] = 256;                                            // tile_dim1 = 256 rows
    g1[5] = stride0;                                        // tensor_dim0_stride (elements)
    g1[6] = 0;
    g1[7] = 0;
    v4i gz4 = {0, 0, 0, 0};
    v8i gz8 = {0, 0, 0, 0, 0, 0, 0, 0};
    __builtin_amdgcn_tensor_load_to_lds(g0, g1, gz4, gz4, gz8, 0);
}
#endif

// One-time fp32 -> f16 conversion of W (transposed) and P into workspace.
__global__ __launch_bounds__(256) void prep_convert_kernel(
    const float* __restrict__ W, const float* __restrict__ P,
    _Float16* __restrict__ Wt16, _Float16* __restrict__ P16)
{
    const int tid = threadIdx.x;
    const int b   = blockIdx.x;
    if (b < DPROJ) {
        // Wt16[n][k] = (f16) W[k][n] ; one n-row per block, 4 k per thread
        const int n  = b;
        const int k0 = tid * 4;
        v4h h;
        #pragma unroll
        for (int i = 0; i < 4; ++i)
            h[i] = (_Float16)W[(size_t)(k0 + i) * DPROJ + n];
        *(v4h*)(Wt16 + (size_t)n * D_IN + k0) = h;
    } else {
        // P16 = (f16) P ; straight copy, coalesced
        const int idx = (b - DPROJ) * 1024 + tid * 4;
        const float4 v = *(const float4*)(P + idx);
        v4h h = {(_Float16)v.x, (_Float16)v.y, (_Float16)v.z, (_Float16)v.w};
        *(v4h*)(P16 + idx) = h;
    }
}

template <bool PREPPED>
__global__ __launch_bounds__(256) void protonet_fused_kernel(
    const float* __restrict__ X,       // [N, 1024]
    const float* __restrict__ mean,    // [1024]
    const float* __restrict__ W,       // [1024, 256]  (used when !PREPPED)
    const float* __restrict__ P,       // [256, 256]
    const _Float16* __restrict__ Wt16, // [256, 1024]  f16 W^T  (PREPPED)
    const _Float16* __restrict__ P16,  // [256, 256]   f16 P    (PREPPED)
    float* __restrict__ out)           // [N, 256]
{
    extern __shared__ unsigned char smem[];
    _Float16* A_lds  = (_Float16*)(smem + OFF_A);
    _Float16* BT0    = (_Float16*)(smem + OFF_BT0);
    _Float16* BT1    = (_Float16*)(smem + OFF_BT1);
    _Float16* Zs     = (_Float16*)(smem + OFF_ZS);
    float*    rowsum = (float*)(smem + OFF_RS);
    float*    p2_lds = (float*)(smem + OFF_P2);

    const int tid  = threadIdx.x;
    const int lane = tid & 31;
    const int wid  = tid >> 5;      // 8 waves
    const int wm   = wid >> 1;      // 0..3 : 16-row strip
    const int wn   = wid & 1;       // 0..1 : 128-col half
    const int rowbase = blockIdx.x * BM;

    if (tid < BM) rowsum[tid] = 0.0f;

    // ---- prototype squared norms in fp32 (P is L2-resident) ----
    {
        const float* prow = P + (size_t)tid * DPROJ;
        float s = 0.0f;
        #pragma unroll 8
        for (int k = 0; k < DPROJ; k += 4) {
            float4 v = *(const float4*)(prow + k);
            s += v.x * v.x + v.y * v.y + v.z * v.z + v.w * v.w;
        }
        p2_lds[tid] = s;
    }

    const int frag_m = wm * 16 + (lane & 15);     // A-fragment row
    const int kb     = (lane >> 4) * 8;           // K sub-offset per half-wave
    const int rbase  = wm * 16 + 8 * (lane >> 4); // first C-row owned by this lane

#if USE_TDM
    const unsigned ldsBT0 = lds_addr_of(BT0);
    const unsigned ldsBT1 = lds_addr_of(BT1);
    if constexpr (PREPPED) {
        if (wid == 0) tdm_load_tile(ldsBT0, Wt16, D_IN, D_IN);   // prologue: tile k0=0
    }
#endif

    // ================= Phase 1: Z = (X - mean) @ W, f16 WMMA, f32 accum ======
    v8f acc[8] = {};
    for (int k0 = 0; k0 < D_IN; k0 += 32) {
        const int buf = (k0 >> 5) & 1;
        _Float16* BTb = buf ? BT1 : BT0;
        __syncthreads();  // prev readers of A and of the other BT buffer are done

        // Stage A tile: (X - mean) -> f16, one ds_store_b128 per thread
        {
            const int m  = tid >> 2;
            const int kk = (tid & 3) * 8;
            const float* src = X + (size_t)(rowbase + m) * D_IN + k0 + kk;
            const float* mu  = mean + k0 + kk;
            const float4 x0 = *(const float4*)(src);
            const float4 x1 = *(const float4*)(src + 4);
            const float4 m0 = *(const float4*)(mu);
            const float4 m1 = *(const float4*)(mu + 4);
            v8h h;
            h[0] = (_Float16)(x0.x - m0.x); h[1] = (_Float16)(x0.y - m0.y);
            h[2] = (_Float16)(x0.z - m0.z); h[3] = (_Float16)(x0.w - m0.w);
            h[4] = (_Float16)(x1.x - m1.x); h[5] = (_Float16)(x1.y - m1.y);
            h[6] = (_Float16)(x1.z - m1.z); h[7] = (_Float16)(x1.w - m1.w);
            *(v8h*)(A_lds + m * LDA + kk) = h;
        }

        if constexpr (PREPPED) {
#if USE_TDM
            if (wid == 0) {
                if (k0 + 32 < D_IN) {   // kick DMA for next tile, ensure current landed
                    tdm_load_tile(buf ? ldsBT0 : ldsBT1, Wt16 + (k0 + 32), D_IN, D_IN);
                    __builtin_amdgcn_s_wait_tensorcnt((short)1);
                } else {
                    __builtin_amdgcn_s_wait_tensorcnt((short)0);
                }
            }
#else
            // vector copy of prepped f16 W^T rows (one row per thread)
            const _Float16* src = Wt16 + (size_t)tid * D_IN + k0;
            _Float16* dst = BTb + tid * LDB;
            #pragma unroll
            for (int i = 0; i < 4; ++i)
                *(v8h*)(dst + i * 8) = *(const v8h*)(src + i * 8);
#endif
        } else {
            // unprepped fallback: convert + transpose-scatter fp32 W
            const int kl = tid >> 3, n0 = (tid & 7) * 32;
            const float* src = W + (size_t)(k0 + kl) * DPROJ + n0;
            #pragma unroll
            for (int i = 0; i < 32; ++i)
                BTb[(n0 + i) * LDB + kl] = (_Float16)src[i];
        }
        __syncthreads();

        v8h alo = *(const v8h*)(A_lds + frag_m * LDA + kb);
        v8h ahi = *(const v8h*)(A_lds + frag_m * LDA + 16 + kb);
        v16h a  = SHUF16(alo, ahi);
        #pragma unroll
        for (int t = 0; t < 8; ++t) {
            const int n = wn * 128 + t * 16 + (lane & 15);
            v8h blo = *(const v8h*)(BTb + n * LDB + kb);
            v8h bhi = *(const v8h*)(BTb + n * LDB + 16 + kb);
            v16h b  = SHUF16(blo, bhi);
            acc[t] = __builtin_amdgcn_wmma_f32_16x16x32_f16(
                false, a, false, b, (short)0, acc[t], false, false);
        }
    }

    // ================= Row L2 norms (LDS atomic reduction) ===================
    {
        float part[8];
        #pragma unroll
        for (int j = 0; j < 8; ++j) part[j] = 0.0f;
        #pragma unroll
        for (int t = 0; t < 8; ++t)
            #pragma unroll
            for (int j = 0; j < 8; ++j) part[j] += acc[t][j] * acc[t][j];
        #pragma unroll
        for (int j = 0; j < 8; ++j) atomicAdd(&rowsum[rbase + j], part[j]);
    }
    __syncthreads();

    float invn[8], z2[8];
    #pragma unroll
    for (int j = 0; j < 8; ++j) {
        const float s   = rowsum[rbase + j];
        const float inv = 1.0f / fmaxf(sqrtf(s), EPS);
        invn[j] = inv;
        z2[j]   = s * inv * inv;   // == ||z_norm||^2 exactly as the reference computes it
    }

    // Normalized Z -> LDS (f16) for the second GEMM
    #pragma unroll
    for (int t = 0; t < 8; ++t) {
        const int c = wn * 128 + t * 16 + (lane & 15);
        #pragma unroll
        for (int j = 0; j < 8; ++j)
            Zs[(rbase + j) * LZS + c] = (_Float16)(acc[t][j] * invn[j]);
    }

#if USE_TDM
    if constexpr (PREPPED) {
        if (wid == 0) tdm_load_tile(ldsBT0, P16, DPROJ, DPROJ);  // phase-2 prologue
    }
#endif

    // ================= Phase 2: dots = Zn @ P^T, f16 WMMA ====================
    v8f acc2[8] = {};
    for (int k0 = 0; k0 < DPROJ; k0 += 32) {
        const int buf = (k0 >> 5) & 1;
        _Float16* BTb = buf ? BT1 : BT0;
        __syncthreads();

        if constexpr (PREPPED) {
#if USE_TDM
            if (wid == 0) {
                if (k0 + 32 < DPROJ) {
                    tdm_load_tile(buf ? ldsBT0 : ldsBT1, P16 + (k0 + 32), DPROJ, DPROJ);
                    __builtin_amdgcn_s_wait_tensorcnt((short)1);
                } else {
                    __builtin_amdgcn_s_wait_tensorcnt((short)0);
                }
            }
#else
            const _Float16* src = P16 + (size_t)tid * DPROJ + k0;
            _Float16* dst = BTb + tid * LDB;
            #pragma unroll
            for (int i = 0; i < 4; ++i)
                *(v8h*)(dst + i * 8) = *(const v8h*)(src + i * 8);
#endif
        } else {
            // unprepped fallback: P is [c][k] row-major already -> direct f16 convert
            const float* src = P + (size_t)tid * DPROJ + k0;
            _Float16* dst = BTb + tid * LDB;
            #pragma unroll
            for (int i = 0; i < 32; ++i) dst[i] = (_Float16)src[i];
        }
        __syncthreads();

        v8h alo = *(const v8h*)(Zs + frag_m * LZS + k0 + kb);
        v8h ahi = *(const v8h*)(Zs + frag_m * LZS + k0 + 16 + kb);
        v16h a  = SHUF16(alo, ahi);
        #pragma unroll
        for (int t = 0; t < 8; ++t) {
            const int n = wn * 128 + t * 16 + (lane & 15);
            v8h blo = *(const v8h*)(BTb + n * LDB + kb);
            v8h bhi = *(const v8h*)(BTb + n * LDB + 16 + kb);
            v16h b  = SHUF16(blo, bhi);
            acc2[t] = __builtin_amdgcn_wmma_f32_16x16x32_f16(
                false, a, false, b, (short)0, acc2[t], false, false);
        }
    }

    // ================= Epilogue: -sqrt(max(z2 + p2 - 2*dot, 0)) ==============
    #pragma unroll
    for (int t = 0; t < 8; ++t) {
        const int c  = wn * 128 + t * 16 + (lane & 15);
        const float pc = p2_lds[c];
        #pragma unroll
        for (int j = 0; j < 8; ++j) {
            float d2 = z2[j] + pc - 2.0f * acc2[t][j];
            d2 = fmaxf(d2, 0.0f);
            out[(size_t)(rowbase + rbase + j) * NCLS + c] = -sqrtf(d2);
        }
    }
}

extern "C" void kernel_launch(void* const* d_in, const int* in_sizes, int n_in,
                              void* d_out, int out_size, void* d_ws, size_t ws_size,
                              hipStream_t stream) {
    const float* X    = (const float*)d_in[0];   // [B,Q,1024] fp32
    const float* mean = (const float*)d_in[1];   // [1024]
    const float* W    = (const float*)d_in[2];   // [1024,256]
    const float* P    = (const float*)d_in[3];   // [256,256]
    float* out        = (float*)d_out;           // [B,Q,256]

    const int N    = in_sizes[0] / D_IN;         // 32768 rows
    const int grid = N / BM;                     // 512 blocks

    const size_t need = (size_t)DPROJ * D_IN * sizeof(_Float16)
                      + (size_t)NCLS * DPROJ * sizeof(_Float16);
    if (ws_size >= need) {
        _Float16* Wt16 = (_Float16*)d_ws;
        _Float16* P16  = Wt16 + (size_t)DPROJ * D_IN;
        prep_convert_kernel<<<DPROJ + (NCLS * DPROJ) / 1024, 256, 0, stream>>>(W, P, Wt16, P16);
        protonet_fused_kernel<true><<<grid, 256, SMEM_BYTES, stream>>>(
            X, mean, W, P, Wt16, P16, out);
    } else {
        protonet_fused_kernel<false><<<grid, 256, SMEM_BYTES, stream>>>(
            X, mean, W, P, nullptr, nullptr, out);
    }
}
